// DepthGuidedSliceDistribution_87668872446272
// MI455X (gfx1250) — compile-verified, matrix operand order checked by reading, not run on hardware
//
#include <hip/hip_runtime.h>
#include <hip/hip_bf16.h>
#include <math.h>

typedef __attribute__((ext_vector_type(16))) _Float16 v16h;
typedef __attribute__((ext_vector_type(8)))  float    v8f;

#define B_   2
#define S_   12
#define C_   128
#define H_   128
#define W_   128
#define HID_ 64
#define HW_  (H_*W_)
#define LD1_ (3*C_)

// A/B operand K mapping for 16-bit WMMA (16x32 tile):
// lanes 0-15 hold K in {0..7, 16..23}, lanes 16-31 hold K in {8..15, 24..31}
__device__ __forceinline__ int kmap(int lane, int e) {
  int kh = (lane >> 4) << 3;           // 0 or 8
  return (e < 8) ? (kh + e) : (16 + kh + (e - 8));
}

// Branchless exact-flavor GELU: erf via Abramowitz-Stegun 7.1.26 (|err|<1.5e-7),
// one v_rcp_f32 + one v_exp_f32 + straight-line FMAs (co-executes with WMMA).
__device__ __forceinline__ float gelu_exact(float x) {
  float ax = fabsf(x) * 0.70710678118654752f;     // |x|/sqrt(2)
  float t  = __builtin_amdgcn_rcpf(1.0f + 0.3275911f * ax);
  float p  = t * (0.254829592f +
             t * (-0.284496736f +
             t * (1.421413741f +
             t * (-1.453152027f +
             t * 1.061405429f))));
  float erf_abs = 1.0f - p * __expf(-ax * ax);
  float erf_x   = copysignf(erf_abs, x);
  return 0.5f * x * (1.0f + erf_x);
}

// Stage a [HID x C] f32 weight matrix into f16 WMMA B-operand tiles in LDS.
__device__ __forceinline__ void stage_w(_Float16* wl, const float* __restrict__ src,
                                        int ld, int tid) {
  for (int i = tid; i < 8192; i += 64) {
    int e  = i & 15;
    int l  = (i >> 4) & 31;
    int t  = i >> 9;                 // kt*4 + nt
    int nt = t & 3, kt = t >> 2;
    int n  = l & 15;
    wl[i] = (_Float16)src[(nt*16 + n) * ld + kt*32 + kmap(l, e)];
  }
}

__launch_bounds__(64)
__global__ void dgsd_fused_kernel(const float* __restrict__ fs,
                                  const float* __restrict__ xrgb,
                                  const float* __restrict__ xdep,
                                  const float* __restrict__ W1,
                                  const float* __restrict__ b1,
                                  const float* __restrict__ W2,
                                  const float* __restrict__ b2,
                                  const float* __restrict__ Wd1,
                                  const float* __restrict__ bd1,
                                  const float* __restrict__ Wd2,
                                  const float* __restrict__ bd2,
                                  const float* __restrict__ taup,
                                  const float* __restrict__ lsig,
                                  const float* __restrict__ lprior,
                                  float* __restrict__ out_xf,
                                  float* __restrict__ out_alpha,
                                  float* __restrict__ out_uf,
                                  float* __restrict__ out_lam)
{
  // One 16KB weight buffer, re-staged per phase: Bd1 -> Br -> Bdp -> Bf.
  __shared__ __attribute__((aligned(32))) _Float16 wl[8192];
  __shared__ __attribute__((aligned(32))) float invl[2048]; // 2 waves x 4 nt x 32 lanes x 8 j
  __shared__ float tl[2 * S_ * 16];                         // per-wave per-slice logits/tau
  __shared__ float bcr[2][16], bce[2][16], bcm[2][16], bcz[2][16];

  const int tid  = threadIdx.x;
  const int wid  = tid >> 5;
  const int lane = tid & 31;
  const int px   = lane & 15;   // pixel within wave tile (A/acc layout)
  const int hi   = lane >> 4;

  const int pixbase = blockIdx.x * 32 + wid * 16;   // 32 pixels/block, 16/wave
  const int b   = pixbase / HW_;
  const int hw0 = pixbase % HW_;

  // scalar params (uniform)
  const float tau_c  = fminf(fmaxf(taup[0], 0.3f), 5.0f);
  const float invtau = 1.0f / tau_c;
  const float sig    = log1pf(__expf(lsig[0])) + 1e-6f;
  const float inv2s2 = 1.0f / (2.0f * sig * sig);
  const float lam    = log1pf(__expf(lprior[0]));

  // ---- load A tiles for x_depth / x_rgb (f32 -> f16, A layout) ----
  v16h ad[4], ar[4];
  {
    const float* xb = xdep + (size_t)b * C_ * HW_ + hw0;
    const float* rb = xrgb + (size_t)b * C_ * HW_ + hw0;
#pragma unroll
    for (int kt = 0; kt < 4; ++kt) {
#pragma unroll
      for (int e = 0; e < 16; ++e) {
        int c = kt * 32 + kmap(lane, e);
        ad[kt][e] = (_Float16)xb[(size_t)c * HW_ + px];
        ar[kt][e] = (_Float16)rb[(size_t)c * HW_ + px];
      }
    }
  }

  // ================= Phase 1: depth head (Wd1) =================
  stage_w(wl, Wd1, C_, tid);
  __syncthreads();

  float mu_j[8];
  {
    float cj[8];
#pragma unroll
    for (int j = 0; j < 8; ++j) cj[j] = 0.f;
#pragma unroll
    for (int nt = 0; nt < 4; ++nt) {
      v8f d = {};
#pragma unroll
      for (int kt = 0; kt < 4; ++kt) {
        const v16h bt = *reinterpret_cast<const v16h*>(&wl[((kt*4 + nt)*32 + lane)*16]);
        d = __builtin_amdgcn_wmma_f32_16x16x32_f16(false, ad[kt], false, bt,
                                                   (short)0, d, false, false);
      }
      float bv = bd1[nt*16 + px];
      float wv = Wd2[nt*16 + px];
#pragma unroll
      for (int j = 0; j < 8; ++j) cj[j] += wv * gelu_exact(d[j] + bv);
    }
#pragma unroll
    for (int j = 0; j < 8; ++j) {
#pragma unroll
      for (int off = 1; off < 16; off <<= 1) cj[j] += __shfl_xor(cj[j], off, 32);
      float dsg = __builtin_amdgcn_rcpf(1.0f + __expf(-(cj[j] + bd2[0])));
      mu_j[j] = (float)(S_ - 1) * dsg;
    }
  }
  __syncthreads();

  // ================= Phase 2: inv, part 1 (Wr) =================
  stage_w(wl, W1 + C_, LD1_, tid);
  __syncthreads();

  v8f dinv[4];
#pragma unroll
  for (int nt = 0; nt < 4; ++nt) {
    v8f d = {};
#pragma unroll
    for (int kt = 0; kt < 4; ++kt) {
      const v16h br = *reinterpret_cast<const v16h*>(&wl[((kt*4 + nt)*32 + lane)*16]);
      d = __builtin_amdgcn_wmma_f32_16x16x32_f16(false, ar[kt], false, br,
                                                 (short)0, d, false, false);
    }
    dinv[nt] = d;
  }
  __syncthreads();

  // ================= Phase 3: inv, part 2 (Wdp) =================
  stage_w(wl, W1 + 2*C_, LD1_, tid);
  __syncthreads();

#pragma unroll
  for (int nt = 0; nt < 4; ++nt) {
    v8f d = dinv[nt];
#pragma unroll
    for (int kt = 0; kt < 4; ++kt) {
      const v16h bp = *reinterpret_cast<const v16h*>(&wl[((kt*4 + nt)*32 + lane)*16]);
      d = __builtin_amdgcn_wmma_f32_16x16x32_f16(false, ad[kt], false, bp,
                                                 (short)0, d, false, false);
    }
    float bv = b1[nt*16 + px];
#pragma unroll
    for (int j = 0; j < 8; ++j)
      invl[((wid*4 + nt)*32 + lane)*8 + j] = d[j] + bv;
  }
  __syncthreads();

  // ================= Phase 4: slice loop (Wf) =================
  stage_w(wl, W1, LD1_, tid);
  __syncthreads();

  float w2v[4];
#pragma unroll
  for (int nt = 0; nt < 4; ++nt) w2v[nt] = W2[nt*16 + px];
  const float b2v = b2[0];

  float m_j[8], z_j[8], ws_j[8];
#pragma unroll
  for (int j = 0; j < 8; ++j) { m_j[j] = -3.0e38f; z_j[j] = 0.f; ws_j[j] = 0.f; }
  float acc[4][16];
#pragma unroll
  for (int kt = 0; kt < 4; ++kt)
#pragma unroll
    for (int e = 0; e < 16; ++e) acc[kt][e] = 0.f;

  const float* fsb = fs + (size_t)b * S_ * C_ * HW_ + hw0;

  for (int s = 0; s < S_; ++s) {
    const float* fss = fsb + (size_t)s * C_ * HW_;
    if (s + 1 < S_) {
      const float* nx = fss + (size_t)C_ * HW_;
      __builtin_prefetch(nx + (size_t)(lane * 4) * HW_ + px, 0, 1);
    }

    // fs tile: 16 px x 128 ch, f32->f16 in A layout (single HBM pass over fs)
    v16h a[4];
#pragma unroll
    for (int kt = 0; kt < 4; ++kt)
#pragma unroll
      for (int e = 0; e < 16; ++e)
        a[kt][e] = (_Float16)fss[(size_t)(kt*32 + kmap(lane, e)) * HW_ + px];

    float cj[8];
#pragma unroll
    for (int j = 0; j < 8; ++j) cj[j] = 0.f;
#pragma unroll
    for (int nt = 0; nt < 4; ++nt) {
      v8f d = *reinterpret_cast<const v8f*>(&invl[((wid*4 + nt)*32 + lane)*8]);
#pragma unroll
      for (int kt = 0; kt < 4; ++kt) {
        const v16h bt = *reinterpret_cast<const v16h*>(&wl[((kt*4 + nt)*32 + lane)*16]);
        d = __builtin_amdgcn_wmma_f32_16x16x32_f16(false, a[kt], false, bt,
                                                   (short)0, d, false, false);
      }
#pragma unroll
      for (int j = 0; j < 8; ++j) cj[j] += w2v[nt] * gelu_exact(d[j]);
    }

#pragma unroll
    for (int j = 0; j < 8; ++j) {
#pragma unroll
      for (int off = 1; off < 16; off <<= 1) cj[j] += __shfl_xor(cj[j], off, 32);
      float dsv = (float)s - mu_j[j];
      float t   = (cj[j] + b2v + lam * (-(dsv*dsv) * inv2s2)) * invtau;
      float mn  = fmaxf(m_j[j], t);
      float r   = __expf(m_j[j] - mn);
      float e   = __expf(t - mn);
      z_j[j]  = z_j[j]  * r + e;
      ws_j[j] = ws_j[j] * r + e * t;
      m_j[j]  = mn;
      if (px == 0) {                 // lanes 0 and 16 own disjoint pixel halves
        int p = j + 8*hi;
        tl[(wid*S_ + s)*16 + p] = t;
        bcr[wid][p] = r;
        bce[wid][p] = e;
      }
    }
    asm volatile("s_wait_dscnt 0" ::: "memory");  // wave-internal LDS broadcast
    float rr = bcr[wid][px];
    float ee = bce[wid][px];
#pragma unroll
    for (int kt = 0; kt < 4; ++kt)
#pragma unroll
      for (int e = 0; e < 16; ++e)
        acc[kt][e] = acc[kt][e] * rr + ee * (float)a[kt][e];
  }

  // ---- finalize: entropy/u_f, stage m/Z, write outputs ----
#pragma unroll
  for (int j = 0; j < 8; ++j) {
    float lnZ = __logf(z_j[j]);
    float ent = m_j[j] + lnZ - ws_j[j] * __builtin_amdgcn_rcpf(z_j[j]);
    if (px == 0) {
      int p = j + 8*hi;
      bcm[wid][p] = m_j[j];
      bcz[wid][p] = z_j[j];
      out_uf[b * HW_ + hw0 + p] = ent * (1.0f / 2.484906649788f);  // 1/ln(12)
    }
  }
  asm volatile("s_wait_dscnt 0" ::: "memory");
  const float invzf = __builtin_amdgcn_rcpf(bcz[wid][px]);

  {
    float* xb = out_xf + (size_t)b * C_ * HW_ + hw0;
#pragma unroll
    for (int kt = 0; kt < 4; ++kt)
#pragma unroll
      for (int e = 0; e < 16; ++e) {
        int c = kt*32 + kmap(lane, e);
        xb[(size_t)c * HW_ + px] = acc[kt][e] * invzf;
      }
  }

  for (int idx = lane; idx < S_*16; idx += 32) {
    int s = idx >> 4;
    int p = idx & 15;
    float t = tl[(wid*S_ + s)*16 + p];
    out_alpha[((size_t)(b*S_ + s)) * HW_ + hw0 + p] =
        __expf(t - bcm[wid][p]) * __builtin_amdgcn_rcpf(bcz[wid][p]);
  }

  if (blockIdx.x == 0 && tid == 0) out_lam[0] = lam;
}

extern "C" void kernel_launch(void* const* d_in, const int* in_sizes, int n_in,
                              void* d_out, int out_size, void* d_ws, size_t ws_size,
                              hipStream_t stream) {
  const float* fs   = (const float*)d_in[0];
  const float* xrgb = (const float*)d_in[1];
  const float* xdep = (const float*)d_in[2];
  const float* W1   = (const float*)d_in[3];
  const float* b1   = (const float*)d_in[4];
  const float* W2   = (const float*)d_in[5];
  const float* b2   = (const float*)d_in[6];
  const float* Wd1  = (const float*)d_in[7];
  const float* bd1  = (const float*)d_in[8];
  const float* Wd2  = (const float*)d_in[9];
  const float* bd2  = (const float*)d_in[10];
  const float* tau  = (const float*)d_in[11];
  const float* lsig = (const float*)d_in[12];
  const float* lpr  = (const float*)d_in[13];

  float* out       = (float*)d_out;
  float* out_xf    = out;
  float* out_alpha = out + (size_t)B_ * C_ * HW_;
  float* out_uf    = out_alpha + (size_t)B_ * S_ * HW_;
  float* out_lam   = out_uf + (size_t)B_ * HW_;

  const int blocks = (B_ * HW_) / 32;   // 1024 blocks, 32 pixels each
  dgsd_fused_kernel<<<blocks, 64, 0, stream>>>(fs, xrgb, xdep, W1, b1, W2, b2,
                                               Wd1, bd1, Wd2, bd2, tau, lsig, lpr,
                                               out_xf, out_alpha, out_uf, out_lam);
}
